// TtBloomAttention_48722109006297
// MI455X (gfx1250) — compile-verified, matrix operand order checked by reading, not run on hardware
//
#include <hip/hip_runtime.h>

// ---------------------------------------------------------------------------
// Bloom attention block for MI455X (gfx1250, wave32, WMMA).
//   B=2, S=2048, H=4096, NH=32, HD=128
// Pipeline:
//   1) qkv_gemm_kernel : fused = hs @ qkv_w^T + qkv_b  -> bf16 Q/K/V (head-major) in ws
//   2) attn_kernel     : flash attention (alibi + mask + online softmax) -> bf16 ctx in ws
//   3) dense_gemm_kernel: out = ctx @ dense_w^T + dense_b + residual (f32)
// All matmuls use v_wmma_f32_16x16x32_bf16 (f32 accumulate).
// ---------------------------------------------------------------------------

#define B_    2
#define S_    2048
#define H_    4096
#define NH_   32
#define HD_   128
#define M_    (B_ * S_)     // 4096 rows
#define NQKV_ 12288         // 3*H
#define INV_NORM_ 0.08838834764831845f  // 1/sqrt(128)
#define BETA_ 1.0f

typedef __attribute__((ext_vector_type(16))) __bf16 v16bf;
typedef __attribute__((ext_vector_type(8)))  __bf16 v8bf;
typedef __attribute__((ext_vector_type(8)))  float  v8f;

// Load a 16x32 WMMA A/B fragment slice for this lane.
// CDNA5 16-bit operand layout: lane L<16 holds K=[8*kh,8*kh+8) in v0..3 and
// K=[16+8*kh, +8) in v4..7 (kh = L>=16). Both slices are 8 contiguous bf16.
static __device__ __forceinline__ v16bf ld_frag(const __bf16* lo_p, const __bf16* hi_p) {
  v8bf lo = *(const v8bf*)lo_p;
  v8bf hi = *(const v8bf*)hi_p;
  v16bf r;
#pragma unroll
  for (int i = 0; i < 8; ++i) { r[i] = lo[i]; r[i + 8] = hi[i]; }
  return r;
}

static __device__ __forceinline__ v8f wmma_bf16(v16bf a, v16bf b, v8f c) {
  // (neg_a, A, neg_b, B, c_mod, C, reuse_a, reuse_b)
  return __builtin_amdgcn_wmma_f32_16x16x32_bf16(false, a, false, b, (short)0, c,
                                                 false, false);
}

// ---------------------------------------------------------------------------
// Kernel 1: QKV projection. Block tile 128x128, 8 waves (4x2), wave tile 32x64.
// A = hidden_states (f32 -> bf16 on stage), B[k][n] = qkv_w[n][k] (k-contiguous).
// Epilogue scatters +bias into head-major bf16 Q/K/V: n = nh*384 + which*128 + hd.
// ---------------------------------------------------------------------------
__global__ __launch_bounds__(256) void qkv_gemm_kernel(
    const float* __restrict__ hs, const float* __restrict__ qkv_w,
    const float* __restrict__ qkv_b,
    __bf16* __restrict__ qw, __bf16* __restrict__ kw, __bf16* __restrict__ vw) {
  __shared__ __align__(16) __bf16 Al[128][32];  // [m][k]
  __shared__ __align__(16) __bf16 Bt[128][32];  // [n][k]

  const int tid  = threadIdx.x;
  const int lane = tid & 31, l16 = lane & 15, kh = lane >> 4;
  const int wv = tid >> 5, wm = wv >> 1, wn = wv & 1;
  const int m0 = blockIdx.y * 128;
  const int n0 = blockIdx.x * 128;

  v8f vzero = {};
  v8f acc[2][4];
#pragma unroll
  for (int i = 0; i < 2; ++i)
#pragma unroll
    for (int j = 0; j < 4; ++j) acc[i][j] = vzero;

  for (int kt = 0; kt < H_; kt += 32) {
#pragma unroll
    for (int it = 0; it < 2; ++it) {
      int g = tid + it * 256;         // 512 groups of 8 elements
      int row = g >> 2;
      int kg = (g & 3) * 8;
      const float* srca = hs + (size_t)(m0 + row) * H_ + kt + kg;
      __bf16* dsta = &Al[row][kg];
#pragma unroll
      for (int e = 0; e < 8; ++e) dsta[e] = (__bf16)srca[e];
      const float* srcb = qkv_w + (size_t)(n0 + row) * H_ + kt + kg;
      __bf16* dstb = &Bt[row][kg];
#pragma unroll
      for (int e = 0; e < 8; ++e) dstb[e] = (__bf16)srcb[e];
    }
    __syncthreads();
#pragma unroll
    for (int mi = 0; mi < 2; ++mi) {
      const __bf16* ar = &Al[wm * 32 + mi * 16 + l16][0];
      v16bf af = ld_frag(ar + 8 * kh, ar + 16 + 8 * kh);
#pragma unroll
      for (int ni = 0; ni < 4; ++ni) {
        const __bf16* br = &Bt[wn * 64 + ni * 16 + l16][0];
        v16bf bf = ld_frag(br + 8 * kh, br + 16 + 8 * kh);
        acc[mi][ni] = wmma_bf16(af, bf, acc[mi][ni]);
      }
    }
    __syncthreads();
  }

#pragma unroll
  for (int mi = 0; mi < 2; ++mi) {
#pragma unroll
    for (int ni = 0; ni < 4; ++ni) {
      int n_base = n0 + wn * 64 + ni * 16;           // 16-tile never crosses q/k/v
      int nh     = n_base / 384;
      int which  = (n_base % 384) / 128;
      int hd     = (n_base % 128) + l16;
      __bf16* dst = which == 0 ? qw : (which == 1 ? kw : vw);
      float bias = qkv_b[n_base + l16];
#pragma unroll
      for (int r = 0; r < 8; ++r) {
        int m_g = m0 + wm * 32 + mi * 16 + r + 8 * kh;
        int b = m_g >> 11, s = m_g & (S_ - 1);
        int head = b * NH_ + nh;
        dst[((size_t)head * S_ + s) * HD_ + hd] = (__bf16)(acc[mi][ni][r] + bias);
      }
    }
  }
}

// ---------------------------------------------------------------------------
// Kernel 2: flash attention. Block = one head x 128 query rows; 8 waves each
// own a 16x128 Q tile (kept in registers as 4 A-fragments). Loop kv in chunks
// of 32: stage K (row-major) and V (transposed) in LDS, QK^T via WMMA, alibi +
// mask + online softmax in f32 (shfl_xor row reductions), P->bf16 via per-wave
// LDS restage, then P@V accumulated into 8 f32 WMMA tiles (16x128).
// ---------------------------------------------------------------------------
__global__ __launch_bounds__(256) void attn_kernel(
    const __bf16* __restrict__ qw, const __bf16* __restrict__ kw,
    const __bf16* __restrict__ vw, const float* __restrict__ alibi,
    const unsigned char* __restrict__ mask, __bf16* __restrict__ ctx) {
  __shared__ __align__(16) __bf16 Kl[32][128];      // [kv][hd]
  __shared__ __align__(16) __bf16 Vt[128][32];      // [hd][kv]
  __shared__ __align__(16) __bf16 Pl[8][16][32];    // per-wave P restage

  const int tid = threadIdx.x;
  const int lane = tid & 31, l16 = lane & 15, kh = lane >> 4;
  const int wv  = tid >> 5;
  const int h   = blockIdx.y;        // 0..B*NH-1
  const int b   = h >> 5;
  const int nh  = h & (NH_ - 1);
  const int qr0 = blockIdx.x * 128 + wv * 16;

  v16bf qf[4];
#pragma unroll
  for (int kk = 0; kk < 4; ++kk) {
    const __bf16* base = qw + ((size_t)h * S_ + qr0 + l16) * HD_ + kk * 32 + 8 * kh;
    qf[kk] = ld_frag(base, base + 16);
  }

  v8f vzero = {};
  v8f o[8];
#pragma unroll
  for (int t = 0; t < 8; ++t) o[t] = vzero;
  float mrow[8], lrow[8];
#pragma unroll
  for (int r = 0; r < 8; ++r) { mrow[r] = -3.0e38f; lrow[r] = 0.0f; }

  for (int kv0 = 0; kv0 < S_; kv0 += 32) {
#pragma unroll
    for (int it = 0; it < 2; ++it) {
      int g  = tid + it * 256;        // 512 groups of 8
      int kv = g >> 4;
      int ko = (g & 15) * 8;
      const __bf16* ksrc = kw + ((size_t)h * S_ + kv0 + kv) * HD_ + ko;
      *(v8bf*)&Kl[kv][ko] = *(const v8bf*)ksrc;
      const __bf16* vsrc = vw + ((size_t)h * S_ + kv0 + kv) * HD_ + ko;
      v8bf vvv = *(const v8bf*)vsrc;
#pragma unroll
      for (int e = 0; e < 8; ++e) Vt[ko + e][kv] = vvv[e];
    }
    __syncthreads();
    if (kv0 + 32 < S_) {  // warm L2/WGP$ for next chunk (global_prefetch_b8)
      size_t nxt = ((size_t)h * S_ + kv0 + 32 + (tid >> 4)) * HD_ + (tid & 15) * 8;
      __builtin_prefetch(kw + nxt, 0, 1);
      __builtin_prefetch(vw + nxt, 0, 1);
    }

    // scores S = Q @ K^T for 16x32 tile (two 16x16 WMMA accumulators)
    v8f c0 = vzero, c1 = vzero;
#pragma unroll
    for (int kk = 0; kk < 4; ++kk) {
      const __bf16* k0 = &Kl[l16][kk * 32];
      c0 = wmma_bf16(qf[kk], ld_frag(k0 + 8 * kh, k0 + 16 + 8 * kh), c0);
      const __bf16* k1 = &Kl[16 + l16][kk * 32];
      c1 = wmma_bf16(qf[kk], ld_frag(k1 + 8 * kh, k1 + 16 + 8 * kh), c1);
    }

    const int col0 = kv0 + l16;
    const int col1 = col0 + 16;
    const float a0 = BETA_ * alibi[(size_t)h * S_ + col0];
    const float a1 = BETA_ * alibi[(size_t)h * S_ + col1];

    float p0[8], p1[8], corr[8];
#pragma unroll
    for (int r = 0; r < 8; ++r) {
      int qrow = qr0 + r + 8 * kh;   // WMMA C layout: VGPR r -> row r + 8*(lane>=16)
      size_t mbase = (size_t)b * S_ * S_ + (size_t)qrow * S_;
      float s0 = c0[r] * INV_NORM_ + a0;
      float s1 = c1[r] * INV_NORM_ + a1;
      if (mask[mbase + col0]) s0 = -1.0e30f;
      if (mask[mbase + col1]) s1 = -1.0e30f;
      // row max over 32 kv columns: pairwise + butterfly across 16-lane half
      float tm = fmaxf(s0, s1);
#pragma unroll
      for (int off = 1; off < 16; off <<= 1) tm = fmaxf(tm, __shfl_xor(tm, off, 32));
      float mn = fmaxf(mrow[r], tm);
      float cr = __expf(mrow[r] - mn);
      float e0 = __expf(s0 - mn);
      float e1 = __expf(s1 - mn);
      float rs = e0 + e1;
#pragma unroll
      for (int off = 1; off < 16; off <<= 1) rs += __shfl_xor(rs, off, 32);
      lrow[r] = lrow[r] * cr + rs;
      mrow[r] = mn;
      corr[r] = cr;
      p0[r] = e0; p1[r] = e1;
    }
#pragma unroll
    for (int t = 0; t < 8; ++t)
#pragma unroll
      for (int r = 0; r < 8; ++r) o[t][r] *= corr[r];

    // restage P through per-wave LDS: C layout -> bf16 A-layout fragment
#pragma unroll
    for (int r = 0; r < 8; ++r) {
      Pl[wv][r + 8 * kh][l16]      = (__bf16)p0[r];
      Pl[wv][r + 8 * kh][16 + l16] = (__bf16)p1[r];
    }
    const __bf16* pr = &Pl[wv][l16][0];
    v16bf pf = ld_frag(pr + 8 * kh, pr + 16 + 8 * kh);
#pragma unroll
    for (int t = 0; t < 8; ++t) {
      const __bf16* vr = &Vt[t * 16 + l16][0];
      o[t] = wmma_bf16(pf, ld_frag(vr + 8 * kh, vr + 16 + 8 * kh), o[t]);
    }
    __syncthreads();
  }

  float inv[8];
#pragma unroll
  for (int r = 0; r < 8; ++r) inv[r] = 1.0f / lrow[r];
#pragma unroll
  for (int t = 0; t < 8; ++t) {
#pragma unroll
    for (int r = 0; r < 8; ++r) {
      int qrow = qr0 + r + 8 * kh;
      int hd = t * 16 + l16;
      ctx[((size_t)b * S_ + qrow) * H_ + nh * HD_ + hd] = (__bf16)(o[t][r] * inv[r]);
    }
  }
}

// ---------------------------------------------------------------------------
// Kernel 3: dense projection + bias + residual. Same tiling as kernel 1;
// A tile loaded directly as bf16 from ctx; epilogue in f32 to d_out.
// ---------------------------------------------------------------------------
__global__ __launch_bounds__(256) void dense_gemm_kernel(
    const __bf16* __restrict__ ctx, const float* __restrict__ dense_w,
    const float* __restrict__ dense_b, const float* __restrict__ residual,
    float* __restrict__ out) {
  __shared__ __align__(16) __bf16 Al[128][32];
  __shared__ __align__(16) __bf16 Bt[128][32];

  const int tid  = threadIdx.x;
  const int lane = tid & 31, l16 = lane & 15, kh = lane >> 4;
  const int wv = tid >> 5, wm = wv >> 1, wn = wv & 1;
  const int m0 = blockIdx.y * 128;
  const int n0 = blockIdx.x * 128;

  v8f vzero = {};
  v8f acc[2][4];
#pragma unroll
  for (int i = 0; i < 2; ++i)
#pragma unroll
    for (int j = 0; j < 4; ++j) acc[i][j] = vzero;

  for (int kt = 0; kt < H_; kt += 32) {
#pragma unroll
    for (int it = 0; it < 2; ++it) {
      int g = tid + it * 256;
      int row = g >> 2;
      int kg = (g & 3) * 8;
      *(v8bf*)&Al[row][kg] =
          *(const v8bf*)(ctx + (size_t)(m0 + row) * H_ + kt + kg);
      const float* srcb = dense_w + (size_t)(n0 + row) * H_ + kt + kg;
      __bf16* dstb = &Bt[row][kg];
#pragma unroll
      for (int e = 0; e < 8; ++e) dstb[e] = (__bf16)srcb[e];
    }
    __syncthreads();
#pragma unroll
    for (int mi = 0; mi < 2; ++mi) {
      const __bf16* ar = &Al[wm * 32 + mi * 16 + l16][0];
      v16bf af = ld_frag(ar + 8 * kh, ar + 16 + 8 * kh);
#pragma unroll
      for (int ni = 0; ni < 4; ++ni) {
        const __bf16* br = &Bt[wn * 64 + ni * 16 + l16][0];
        v16bf bf = ld_frag(br + 8 * kh, br + 16 + 8 * kh);
        acc[mi][ni] = wmma_bf16(af, bf, acc[mi][ni]);
      }
    }
    __syncthreads();
  }

#pragma unroll
  for (int mi = 0; mi < 2; ++mi) {
#pragma unroll
    for (int ni = 0; ni < 4; ++ni) {
      int n_g = n0 + wn * 64 + ni * 16 + l16;
      float bias = dense_b[n_g];
#pragma unroll
      for (int r = 0; r < 8; ++r) {
        int m_g = m0 + wm * 32 + mi * 16 + r + 8 * kh;
        size_t idx = (size_t)m_g * H_ + n_g;
        out[idx] = acc[mi][ni][r] + bias + residual[idx];
      }
    }
  }
}

// ---------------------------------------------------------------------------
extern "C" void kernel_launch(void* const* d_in, const int* in_sizes, int n_in,
                              void* d_out, int out_size, void* d_ws, size_t ws_size,
                              hipStream_t stream) {
  const float* hs        = (const float*)d_in[0];
  const float* residual  = (const float*)d_in[1];
  const float* alibi     = (const float*)d_in[2];
  const unsigned char* mask = (const unsigned char*)d_in[3];  // bool array
  const float* qkv_w     = (const float*)d_in[4];
  const float* qkv_b     = (const float*)d_in[5];
  const float* dense_w   = (const float*)d_in[6];
  const float* dense_b   = (const float*)d_in[7];
  float* out             = (float*)d_out;

  const size_t NQ = (size_t)B_ * NH_ * S_ * HD_;   // 16,777,216 elems
  __bf16* qw  = (__bf16*)d_ws;                     // bf16 Q  (B*NH, S, HD)
  __bf16* kw  = qw + NQ;                           // bf16 K  (B*NH, S, HD)
  __bf16* vw  = kw + NQ;                           // bf16 V  (B*NH, S, HD)
  __bf16* ctx = vw + NQ;                           // bf16 ctx (B, S, H)

  dim3 blk(256);
  qkv_gemm_kernel<<<dim3(NQKV_ / 128, M_ / 128), blk, 0, stream>>>(
      hs, qkv_w, qkv_b, qw, kw, vw);
  attn_kernel<<<dim3(S_ / 128, B_ * NH_), blk, 0, stream>>>(
      qw, kw, vw, alibi, mask, ctx);
  dense_gemm_kernel<<<dim3(H_ / 128, M_ / 128), blk, 0, stream>>>(
      ctx, dense_w, dense_b, residual, out);
}
